// NonLocalMSA_21466246545705
// MI455X (gfx1250) — matrix operand order repaired and checked by reading, not compiled, not run on hardware
//
#include <hip/hip_runtime.h>
#include <hip/hip_bf16.h>

// ---- CDNA5 WMMA fragment types (wave32) ----
typedef __attribute__((ext_vector_type(16))) __bf16        v16bf;
typedef __attribute__((ext_vector_type(8)))  float         v8f;
typedef __attribute__((ext_vector_type(8)))  unsigned int  v8u;
typedef __attribute__((ext_vector_type(4)))  unsigned int  v4u;
typedef __attribute__((ext_vector_type(4)))  float         v4f;

// Geometry
#define B_   16
#define C_   64
#define H_   256
#define W_   256
#define HW_  65536
#define NWIN 64        // 8x8 windows = tokens
#define HD_  8192      // head dim
#define SCALE 0.011048543456039806f   // 1/sqrt(8192)

// ISA 7.12.2 bf16 16x16x32 fragment layouts (wave32):
//  A: lane holds row m=lane%16; K idx for element e: ((e<8)?e:e+8) + 8*(lane>=16)
//     -> per lane: two contiguous 8-element (16B) runs  => 2 x b128 loads
//  B: lane holds col n=lane%16; K idx for element e: ((e<8)?e:e+8) + 8*(lane>=16)
//     -> per lane: two contiguous 8-element (16B) runs  => 2 x b128 loads
//  C/D f32: element r -> m = r + 8*(lane>=16), n = lane%16

static __device__ inline v16bf frag_join(v4u lo, v4u hi) {
  v8u u;
  u[0] = lo[0]; u[1] = lo[1]; u[2] = lo[2]; u[3] = lo[3];
  u[4] = hi[0]; u[5] = hi[1]; u[6] = hi[2]; u[7] = hi[3];
  return __builtin_bit_cast(v16bf, u);
}

// Load both 32-K A fragments of one 16-row weight tile (f32 weights) with
// 8 x float4 (b128) loads, converting to bf16 in registers.
static __device__ inline void load_w_frags(const v4f* w4row, int hi,
                                           v16bf& a0, v16bf& a1) {
  const v4f c0 = w4row[2 * hi],      c1 = w4row[2 * hi + 1];
  const v4f c2 = w4row[4 + 2 * hi],  c3 = w4row[4 + 2 * hi + 1];
  const v4f c4 = w4row[8 + 2 * hi],  c5 = w4row[8 + 2 * hi + 1];
  const v4f c6 = w4row[12 + 2 * hi], c7 = w4row[12 + 2 * hi + 1];
#pragma unroll
  for (int e = 0; e < 4; ++e) {
    a0[e]      = (__bf16)c0[e];  a0[4 + e]  = (__bf16)c1[e];
    a0[8 + e]  = (__bf16)c2[e];  a0[12 + e] = (__bf16)c3[e];
    a1[e]      = (__bf16)c4[e];  a1[4 + e]  = (__bf16)c5[e];
    a1[8 + e]  = (__bf16)c6[e];  a1[12 + e] = (__bf16)c7[e];
  }
}

// ============================================================
// Kernel 1: qkv = w_qkv (192x64) @ x (64 x HW), bf16 out.
// One block = 128 pixels x ALL 192 out-channels (12 M-tiles share the
// B fragments -> x is read from HBM exactly once).
// grid(512, 16 batch) block(256) = 8 waves x 16 pixel-cols.
// ============================================================
__global__ void __launch_bounds__(256) qkv_gemm(const float* __restrict__ x,
                                                const float* __restrict__ w,
                                                __bf16* __restrict__ qkv) {
  const int lane = threadIdx.x & 31;
  const int wave = threadIdx.x >> 5;
  const int col  = lane & 15;
  const int hi   = lane >> 4;
  const int nIdx = blockIdx.x * 128 + wave * 16 + col;
  const int b    = blockIdx.y;
  const float* xb = x + (size_t)b * C_ * HW_;

  // B fragments for K = 0..31 and 32..63 (pixel columns)
  v16bf bf0, bf1;
#pragma unroll
  for (int e = 0; e < 16; ++e) {
    const int k = ((e < 8) ? e : e + 8) + hi * 8;
    bf0[e] = (__bf16)xb[(size_t)k * HW_ + nIdx];
    bf1[e] = (__bf16)xb[(size_t)(32 + k) * HW_ + nIdx];
  }

  v8f acc[12] = {};
  const v4f* w4 = (const v4f*)w;           // 16 float4 per 64-f32 row
#pragma unroll
  for (int mt = 0; mt < 12; ++mt) {
    v16bf a0, a1;
    load_w_frags(w4 + (size_t)(mt * 16 + col) * 16, hi, a0, a1);
    acc[mt] = __builtin_amdgcn_wmma_f32_16x16x32_bf16(false, a0, false, bf0,
                                                      (short)0, acc[mt], false, false);
    acc[mt] = __builtin_amdgcn_wmma_f32_16x16x32_bf16(false, a1, false, bf1,
                                                      (short)0, acc[mt], false, false);
  }

  __bf16* ob = qkv + (size_t)b * 192 * HW_;
#pragma unroll
  for (int mt = 0; mt < 12; ++mt)
#pragma unroll
    for (int r = 0; r < 8; ++r) {
      const int o = mt * 16 + r + hi * 8;
      ob[(size_t)o * HW_ + nIdx] = (__bf16)acc[mt][r];
    }
}

// ============================================================
// Kernel 2: depthwise 3x3 (pad 1) via LDS strip + scatter into token layout
//   q,k : [b*8+h][token][d]  bf16   (d = (c*4+r)*32 + b1l)
//   v   : [b*8+h][d][token]  bf16   (transposed for B fragments)
// grid(64 y-strips, 192 ch, 16 b) block(256 x); 4 output rows per block
// from a 6x258 zero-padded LDS tile (1.5x read amplification vs 9x).
// ============================================================
__global__ void __launch_bounds__(256) dwconv_scatter(const __bf16* __restrict__ qkv,
                                                      const float* __restrict__ wdw,
                                                      __bf16* __restrict__ qtok,
                                                      __bf16* __restrict__ ktok,
                                                      __bf16* __restrict__ vtok) {
  __shared__ float tile[6][258];
  const int xp = threadIdx.x;
  const int y0 = blockIdx.x * 4;
  const int ch = blockIdx.y;
  const int b  = blockIdx.z;
  const __bf16* p = qkv + (size_t)(b * 192 + ch) * HW_;

#pragma unroll
  for (int r = 0; r < 6; ++r) {
    const int yy = y0 - 1 + r;
    tile[r][xp + 1] = (yy >= 0 && yy < H_) ? (float)p[yy * W_ + xp] : 0.f;
  }
  if (xp < 6)    tile[xp][0] = 0.f;
  if (xp >= 250) tile[xp - 250][257] = 0.f;
  __syncthreads();

  float wk[9];
#pragma unroll
  for (int i = 0; i < 9; ++i) wk[i] = wdw[ch * 9 + i];

  const int part = ch >> 6, c = ch & 63;
  const int wx = xp >> 5, b1l = xp & 31;

#pragma unroll
  for (int dy = 0; dy < 4; ++dy) {
    float s = 0.f;
#pragma unroll
    for (int ky = 0; ky < 3; ++ky)
#pragma unroll
      for (int kx = 0; kx < 3; ++kx)
        s += wk[ky * 3 + kx] * tile[dy + ky][xp + kx];

    const int y  = y0 + dy;
    const int wy = y >> 5, b0l = y & 31;
    const int h  = b0l >> 2, r = b0l & 3;
    const int t  = wy * 8 + wx;
    const int d  = (c * 4 + r) * 32 + b1l;
    const int bh = b * 8 + h;
    const __bf16 v = (__bf16)s;
    if (part == 0)      qtok[((size_t)bh * NWIN + t) * HD_ + d] = v;
    else if (part == 1) ktok[((size_t)bh * NWIN + t) * HD_ + d] = v;
    else                vtok[((size_t)bh * HD_ + d) * NWIN + t] = v;
  }
}

// ============================================================
// Kernel 3: sim = scale*(q @ k^T) + pos ; row softmax ; attn bf16.
// One block per (b,h): 512 threads = 16 waves, one 16x16 sim tile each,
// K-loop over 8192 in 32-steps; fragments are 2 x b128 per operand.
// ============================================================
__global__ void __launch_bounds__(512) attn_scores(const __bf16* __restrict__ qtok,
                                                   const __bf16* __restrict__ ktok,
                                                   const float* __restrict__ pos,
                                                   __bf16* __restrict__ attn) {
  __shared__ float sim[NWIN * NWIN];   // 16 KB
  const int lane = threadIdx.x & 31;
  const int wave = threadIdx.x >> 5;
  const int col  = lane & 15;
  const int hi   = lane >> 4;
  const int bh   = blockIdx.x;
  const int h    = bh & 7;
  const int mT   = wave >> 2;
  const int nT   = wave & 3;

  const __bf16* qrow = qtok + ((size_t)bh * NWIN + mT * 16 + col) * HD_;
  const __bf16* krow = ktok + ((size_t)bh * NWIN + nT * 16 + col) * HD_;
  const v4u* q16 = (const v4u*)qrow;   // 16-byte units
  const v4u* k16 = (const v4u*)krow;

  v8f acc = {};
  for (int t = 0; t < HD_ / 32; ++t) {         // kb = 32*t
    // A (q): chunks at 16B indices 4t+hi and 4t+hi+2
    const v4u alo = q16[4 * t + hi];
    const v4u ahi = q16[4 * t + hi + 2];
    // B (k): chunks at 16B indices 4t+hi and 4t+hi+2
    const v4u blo = k16[4 * t + hi];
    const v4u bhi = k16[4 * t + hi + 2];
    if (t + 8 < HD_ / 32) {                    // stream prefetch (global_prefetch_b8)
      __builtin_prefetch((const char*)(qrow + 32 * t + 256), 0, 1);
      __builtin_prefetch((const char*)(krow + 32 * t + 256), 0, 1);
    }
    acc = __builtin_amdgcn_wmma_f32_16x16x32_bf16(false, frag_join(alo, ahi),
                                                  false, frag_join(blo, bhi),
                                                  (short)0, acc, false, false);
  }

#pragma unroll
  for (int r = 0; r < 8; ++r) {
    const int m = mT * 16 + r + hi * 8;
    const int n = nT * 16 + col;
    sim[m * NWIN + n] = acc[r] * SCALE + pos[(h * NWIN + m) * NWIN + n];
  }
  __syncthreads();

  if (threadIdx.x < NWIN) {
    const int m = threadIdx.x;
    const float* row = &sim[m * NWIN];
    float mx = row[0];
    for (int j = 1; j < NWIN; ++j) mx = fmaxf(mx, row[j]);
    float sum = 0.f;
    for (int j = 0; j < NWIN; ++j) sum += __expf(row[j] - mx);
    const float inv = 1.f / sum;
    __bf16* arow = attn + ((size_t)bh * NWIN + m) * NWIN;
    for (int j = 0; j < NWIN; ++j) arow[j] = (__bf16)(__expf(row[j] - mx) * inv);
  }
}

// ============================================================
// Kernel 4: out = attn @ v (M=64 tokens, K=64, N=8192), b128 fragment
// loads, scatter bf16 back to [b][c][y][x].
// grid(64 dChunks, 8 h, 16 b) block(256): 8 waves x 16 d'-cols; each wave
// computes all 4 M-tiles reusing its B fragment.
// ============================================================
__global__ void __launch_bounds__(256) attn_apply(const __bf16* __restrict__ attn,
                                                  const __bf16* __restrict__ vtok,
                                                  __bf16* __restrict__ ao) {
  const int lane = threadIdx.x & 31;
  const int wave = threadIdx.x >> 5;
  const int col  = lane & 15;
  const int hi   = lane >> 4;
  const int h = blockIdx.y, b = blockIdx.z;
  const int bh = b * 8 + h;
  const int n  = blockIdx.x * 128 + wave * 16 + col;     // d' column

  const v4u* v16 = (const v4u*)(vtok + ((size_t)bh * HD_ + n) * NWIN);
  const __bf16* at = attn + (size_t)bh * NWIN * NWIN;

  v8f acc[4] = {};
#pragma unroll
  for (int kt = 0; kt < 2; ++kt) {
    const v16bf bb = frag_join(v16[4 * kt + hi], v16[4 * kt + hi + 2]);
#pragma unroll
    for (int mt = 0; mt < 4; ++mt) {
      const v4u* a16 = (const v4u*)(at + (size_t)(mt * 16 + col) * NWIN);
      const v16bf aa = frag_join(a16[4 * kt + hi], a16[4 * kt + hi + 2]);
      acc[mt] = __builtin_amdgcn_wmma_f32_16x16x32_bf16(false, aa, false, bb,
                                                        (short)0, acc[mt], false, false);
    }
  }

  // d' = (c*4 + r2)*32 + b1l
  const int c = n >> 7, r2 = (n >> 5) & 3, b1l = n & 31;
#pragma unroll
  for (int mt = 0; mt < 4; ++mt)
#pragma unroll
    for (int r = 0; r < 8; ++r) {
      const int t  = mt * 16 + r + hi * 8;               // token
      const int y  = (t >> 3) * 32 + h * 4 + r2;
      const int xx = (t & 7) * 32 + b1l;
      ao[(size_t)(b * C_ + c) * HW_ + y * W_ + xx] = (__bf16)acc[mt][r];
    }
}

// ============================================================
// Kernel 5: out = w_proj (64x64) @ ao + bias, fp32 out.
// All 4 M-tiles per block share the B fragments (ao read once).
// grid(512, 16 batch) block(256).
// ============================================================
__global__ void __launch_bounds__(256) proj_gemm(const __bf16* __restrict__ ao,
                                                 const float* __restrict__ wp,
                                                 const float* __restrict__ bias,
                                                 float* __restrict__ out) {
  const int lane = threadIdx.x & 31;
  const int wave = threadIdx.x >> 5;
  const int col  = lane & 15;
  const int hi   = lane >> 4;
  const int nIdx = blockIdx.x * 128 + wave * 16 + col;
  const int b    = blockIdx.y;
  const __bf16* ab = ao + (size_t)b * C_ * HW_;

  v16bf bf0, bf1;
#pragma unroll
  for (int e = 0; e < 16; ++e) {
    const int k = ((e < 8) ? e : e + 8) + hi * 8;
    bf0[e] = ab[(size_t)k * HW_ + nIdx];
    bf1[e] = ab[(size_t)(32 + k) * HW_ + nIdx];
  }

  v8f acc[4] = {};
  const v4f* w4 = (const v4f*)wp;
#pragma unroll
  for (int mt = 0; mt < 4; ++mt) {
    v16bf a0, a1;
    load_w_frags(w4 + (size_t)(mt * 16 + col) * 16, hi, a0, a1);
    acc[mt] = __builtin_amdgcn_wmma_f32_16x16x32_bf16(false, a0, false, bf0,
                                                      (short)0, acc[mt], false, false);
    acc[mt] = __builtin_amdgcn_wmma_f32_16x16x32_bf16(false, a1, false, bf1,
                                                      (short)0, acc[mt], false, false);
  }

  float* ob = out + (size_t)b * C_ * HW_;
#pragma unroll
  for (int mt = 0; mt < 4; ++mt)
#pragma unroll
    for (int r = 0; r < 8; ++r) {
      const int o = mt * 16 + r + hi * 8;
      ob[(size_t)o * HW_ + nIdx] = acc[mt][r] + bias[o];
    }
}

// ============================================================
extern "C" void kernel_launch(void* const* d_in, const int* in_sizes, int n_in,
                              void* d_out, int out_size, void* d_ws, size_t ws_size,
                              hipStream_t stream) {
  (void)in_sizes; (void)n_in; (void)out_size; (void)ws_size;
  const float* x      = (const float*)d_in[0];
  const float* w_qkv  = (const float*)d_in[1];
  const float* w_dw   = (const float*)d_in[2];
  const float* w_proj = (const float*)d_in[3];
  const float* b_proj = (const float*)d_in[4];
  const float* pos    = (const float*)d_in[5];
  float* out = (float*)d_out;

  // Workspace (total ~806 MB):
  //   [0)          qkv bf16   16*192*65536*2 = 402,653,184 B  (freed after dwconv)
  //   [402653184)  qtok bf16  134,217,728 B
  //   [+134M)      ktok bf16  134,217,728 B
  //   [+134M)      vtok bf16  134,217,728 B  ([d][token] transposed)
  //   [+134M)      attn bf16    1,048,576 B
  //   ao bf16 aliases the qkv region.
  char* ws = (char*)d_ws;
  __bf16* qkvb = (__bf16*)ws;
  __bf16* qtok = (__bf16*)(ws + 402653184ull);
  __bf16* ktok = qtok + 67108864ull;
  __bf16* vtok = ktok + 67108864ull;
  __bf16* attn = vtok + 67108864ull;
  __bf16* ao   = (__bf16*)ws;          // reuse

  qkv_gemm<<<dim3(512, 16), 256, 0, stream>>>(x, w_qkv, qkvb);
  dwconv_scatter<<<dim3(64, 192, 16), 256, 0, stream>>>(qkvb, w_dw, qtok, ktok, vtok);
  attn_scores<<<dim3(128), 512, 0, stream>>>(qtok, ktok, pos, attn);
  attn_apply<<<dim3(64, 8, 16), 256, 0, stream>>>(attn, vtok, ao);
  proj_gemm<<<dim3(512, 16), 256, 0, stream>>>(ao, w_proj, b_proj, out);
}
